// Edge_Predictor_41042707481006
// MI455X (gfx1250) — matrix operand order repaired and checked by reading, not compiled, not run on hardware
//
#include <hip/hip_runtime.h>
#include <hip/hip_bf16.h>
#include <math.h>

typedef __attribute__((ext_vector_type(16))) __bf16 v16bf;
typedef __attribute__((ext_vector_type(8)))  float  v8f;

#define DFEAT   128
#define HID     128
#define NT_N    8   // HID / 16
#define KSTEPS  8   // (2*DFEAT) / 32

union V16U8 { v16bf v; unsigned int u[8]; };

// packed f32x2 -> bf16x2 (v_cvt_pk_bf16_f32), lo -> bits [15:0]
__device__ __forceinline__ unsigned int pack2_bf16(float lo, float hi) {
  union { __hip_bfloat162 b; unsigned int u; } cvt;
  cvt.b = __float22bfloat162_rn(float2{lo, hi});
  return cvt.u;
}

// branch-free tanh: 1 - 2/(1+exp(2x)); saturates correctly at +-inf
__device__ __forceinline__ float fast_tanh(float x) {
  return 1.0f - 2.0f / (1.0f + __expf(2.0f * x));
}

__global__ __launch_bounds__(256, 1) void Edge_Predictor_kernel(
    const float* __restrict__ x,      // [N_NODES, 128] f32
    const int*   __restrict__ eidx,   // [2, E] int32
    const float* __restrict__ W1,     // [256, 128] f32 row-major
    const float* __restrict__ b1,     // [128]
    const float* __restrict__ W2,     // [128, 1]
    const float* __restrict__ b2,     // [1]
    float*       __restrict__ out,    // [E]
    int n_edges)
{
  // W1 in bf16, pre-swizzled to WMMA B-matrix layout:
  // Bmat[(n*8 + kstep)*32 + lane] is the v16bf this lane feeds to wmma.
  __shared__ V16U8 Bmat[NT_N * KSTEPS * 32];   // 64 KB

  const int tid = threadIdx.x;

  // ---- cooperative fill: W1 (f32) -> LDS (bf16, B layout) ----
  {
    unsigned int* lds_u = (unsigned int*)Bmat;
    for (int uidx = tid; uidx < NT_N * KSTEPS * 32 * 8; uidx += 256) {
      int j   = uidx & 7;          // dword within lane's 8-dword chunk
      int L   = (uidx >> 3) & 31;  // lane
      int g   = uidx >> 8;         // group = n*8 + s
      int s   = g & 7;             // K-step
      int n   = g >> 3;            // N-tile
      int col = n * 16 + (L & 15);                  // B column (hidden idx)
      int k0  = s * 32 + ((L >> 4) << 4) + 2 * j;   // lanes 0-15: K 0-15; 16-31: K 16-31
      lds_u[uidx] = pack2_bf16(W1[k0 * HID + col], W1[(k0 + 1) * HID + col]);
    }
  }
  __syncthreads();

  const int lane  = tid & 31;
  const int khalf = lane >> 4;       // 0 = lanes 0-15, 1 = lanes 16-31
  const int m15   = lane & 15;
  const int nwaves = gridDim.x * (blockDim.x >> 5);
  const int wave   = blockIdx.x * (blockDim.x >> 5) + (tid >> 5);
  const int ntiles = (n_edges + 15) >> 4;
  const float b2v  = b2[0];

  // tile-invariant epilogue constants: b1 / W2 for this lane's column
  float b1r[NT_N], w2r[NT_N];
#pragma unroll
  for (int n = 0; n < NT_N; ++n) {
    b1r[n] = b1[n * 16 + m15];
    w2r[n] = W2[n * 16 + m15];
  }

  for (int tile = wave; tile < ntiles; tile += nwaves) {
    int e  = tile * 16 + m15;                 // this lane's edge row (M = m15)
    int ec = e < n_edges ? e : n_edges - 1;
    const float* srcp = x + (long)eidx[ec]           * DFEAT;
    const float* dstp = x + (long)eidx[n_edges + ec] * DFEAT;

    // ---- gather + convert A tiles (16 edges x 32 K per step, bf16) ----
    // A layout: lanes 0-15 hold K {0..15}, lanes 16-31 hold K {16..31}
    V16U8 a[KSTEPS];
#pragma unroll
    for (int kb = 0; kb < KSTEPS; ++kb) {
      const float* rowp = (kb < 4) ? srcp : dstp;   // concat boundary at K=128
      int koff = (kb & 3) * 32 + khalf * 16;
      const float4* p0 = (const float4*)(rowp + koff);
      const float4* p1 = (const float4*)(rowp + koff + 8);
      float4 f0 = p0[0], f1 = p0[1];
      float4 g0 = p1[0], g1 = p1[1];
      a[kb].u[0] = pack2_bf16(f0.x, f0.y);
      a[kb].u[1] = pack2_bf16(f0.z, f0.w);
      a[kb].u[2] = pack2_bf16(f1.x, f1.y);
      a[kb].u[3] = pack2_bf16(f1.z, f1.w);
      a[kb].u[4] = pack2_bf16(g0.x, g0.y);
      a[kb].u[5] = pack2_bf16(g0.z, g0.w);
      a[kb].u[6] = pack2_bf16(g1.x, g1.y);
      a[kb].u[7] = pack2_bf16(g1.z, g1.w);
    }

    float partial[8];
#pragma unroll
    for (int r = 0; r < 8; ++r) partial[r] = 0.0f;

    // ---- GEMM: one 16-wide hidden tile at a time (keep B pressure low) ----
#pragma unroll 1
    for (int n = 0; n < NT_N; ++n) {
      v8f acc = {};
#pragma unroll
      for (int kb = 0; kb < KSTEPS; ++kb) {
        v16bf bm = Bmat[(n * KSTEPS + kb) * 32 + lane].v;
        acc = __builtin_amdgcn_wmma_f32_16x16x32_bf16(
            false, a[kb].v, false, bm, (short)0, acc, false, false);
      }
      // C layout: lane = column N (hidden), VGPR r = row M = r + 8*khalf (edge)
      float b1v = b1r[n];
      float w2v = w2r[n];
#pragma unroll
      for (int r = 0; r < 8; ++r) {
        float h = fast_tanh(acc[r] + b1v);
        partial[r] += h * w2v;     // fold second GEMM (HID x 1) into epilogue
      }
    }

    // ---- reduce the 16 hidden-columns per edge row (within each half-wave) ----
#pragma unroll
    for (int r = 0; r < 8; ++r) {
      float p = partial[r];
#pragma unroll
      for (int off = 8; off >= 1; off >>= 1)
        p += __shfl_xor(p, off, 16);
      partial[r] = p;
    }

    if (m15 == 0) {
      int base = tile * 16 + khalf * 8;
#pragma unroll
      for (int r = 0; r < 8; ++r) {
        int o = base + r;
        if (o < n_edges)
          out[o] = 1.0f / (1.0f + __expf(-(partial[r] + b2v)));
      }
    }
  }
}

extern "C" void kernel_launch(void* const* d_in, const int* in_sizes, int n_in,
                              void* d_out, int out_size, void* d_ws, size_t ws_size,
                              hipStream_t stream) {
  (void)n_in; (void)d_ws; (void)ws_size; (void)out_size;
  const float* x    = (const float*)d_in[0];
  const int*   eidx = (const int*)  d_in[1];   // [2, E] int32 (JAX default x64-off)
  const float* W1   = (const float*)d_in[2];
  const float* b1   = (const float*)d_in[3];
  const float* W2   = (const float*)d_in[4];
  const float* b2   = (const float*)d_in[5];
  float* out = (float*)d_out;

  int n_edges = in_sizes[1] / 2;

  dim3 block(256);          // 8 waves per block
  dim3 grid(512);           // 4096 waves grid-striding over 31250 tiles
  Edge_Predictor_kernel<<<grid, block, 0, stream>>>(x, eidx, W1, b1, W2, b2,
                                                    out, n_edges);
}
